// BiLSTMCrf_69501160784641
// MI455X (gfx1250) — matrix operand order repaired and checked by reading, not compiled, not run on hardware
//
#include <hip/hip_runtime.h>
#include <hip/hip_bf16.h>
#include <stdint.h>

// Problem constants (from reference): T=512, B=512, H=512, K=48
#define TT 512
#define BB 512
#define HH 512
#define KK 48

typedef __attribute__((ext_vector_type(16))) __bf16 bf16x16;
typedef __attribute__((ext_vector_type(8)))  float  f32x8;
typedef __attribute__((ext_vector_type(4)))  float  f32x4;

// ---------------------------------------------------------------------------
// Phase 1: emissions[t,b,k] = hidden[t,b,:] @ W[:,k] + bias[k]
// Memory-bound GEMM (512 MB of `hidden` streamed once, ~22us at 23.3 TB/s).
// bf16 WMMA with f32 accumulation. Each workgroup: 8 waves; W pre-packed into
// 48 KB of LDS as WMMA B-fragments (16 k-steps x 3 n-tiles x 32 lanes x 16
// bf16). Each wave owns a 32-row super-tile (two 16-row WMMA tiles) so every
// B fragment loaded from LDS feeds TWO WMMAs -> half the ds traffic per WMMA
// and 6 WMMAs per k-step to hide A-load latency.
// ---------------------------------------------------------------------------
__global__ __launch_bounds__(256)
void emissions_gemm_kernel(const float* __restrict__ hidden,
                           const float* __restrict__ W,
                           const float* __restrict__ bias,
                           float* __restrict__ emis)
{
    // B-fragments for all (k-step, n-tile): [ (kt*3+nt)*32 + lane ]
    __shared__ bf16x16 sB[16 * 3 * 32];   // 48 KB

    const int tid  = threadIdx.x;
    const int lane = tid & 31;
    const int wave = tid >> 5;
    const int hi   = lane >> 4;      // lane half (0/1)
    const int lidx = lane & 15;

    // ---- Prologue: pack W (fp32, HxK row-major) -> bf16 B-fragments in LDS.
    // B 32x16 (KxN) 16-bit layout: lanes 0-15 hold cols n=0..15, K=0..7,16..23
    // (2 per VGPR in order); lanes 16-31 hold same cols, K=8..15,24..31.
    for (int r = 0; r < 6; ++r) {
        int idx = tid + r * 256;           // 0..1535
        int f   = idx >> 5;                // fragment id 0..47
        int l   = idx & 31;
        int kt  = f / 3;
        int nt  = f - kt * 3;
        int lhi = l >> 4;
        int n   = nt * 16 + (l & 15);
        int hb  = kt * 32 + lhi * 8;       // K = hb+0..7, hb+16..23 for this lane
        bf16x16 vb;
#pragma unroll
        for (int j = 0; j < 16; ++j)
            vb[j] = (__bf16)W[(size_t)(hb + (j < 8 ? j : j + 8)) * KK + n];
        sB[idx] = vb;
    }
    __syncthreads();

    const int npairs = (TT * BB) / 32;                 // 8192 super-tiles
    const int nwaves = gridDim.x * 8;
    const float bi0 = bias[lidx];
    const float bi1 = bias[16 + lidx];
    const float bi2 = bias[32 + lidx];

    for (int pair = blockIdx.x * 8 + wave; pair < npairs; pair += nwaves) {
        const float* arow0 = hidden + (size_t)(pair * 32 + lidx) * HH;
        const float* arow1 = arow0 + (size_t)16 * HH;

        // Prefetch the next super-tile's rows (global_prefetch_b8).
        int np2 = pair + nwaves;
        if (np2 < npairs)
            __builtin_prefetch(hidden + (size_t)(np2 * 32 + lidx) * HH, 0, 1);

        f32x8 c00 = {}, c01 = {}, c02 = {};   // tile 0, n-tiles 0..2
        f32x8 c10 = {}, c11 = {}, c12 = {};   // tile 1, n-tiles 0..2
#pragma unroll 2
        for (int kt = 0; kt < 16; ++kt) {
            // A 16x32 (MxK) 16-bit layout: lane row m=lidx; elements
            // K = { hi*8+0..7, 16+hi*8+0..7 } -> two contiguous f32x4 pairs.
            int hb = kt * 32 + hi * 8;
            f32x4 a00 = *(const f32x4*)(arow0 + hb);
            f32x4 a01 = *(const f32x4*)(arow0 + hb + 4);
            f32x4 a02 = *(const f32x4*)(arow0 + hb + 16);
            f32x4 a03 = *(const f32x4*)(arow0 + hb + 20);
            f32x4 a10 = *(const f32x4*)(arow1 + hb);
            f32x4 a11 = *(const f32x4*)(arow1 + hb + 4);
            f32x4 a12 = *(const f32x4*)(arow1 + hb + 16);
            f32x4 a13 = *(const f32x4*)(arow1 + hb + 20);
            bf16x16 a0, a1;
#pragma unroll
            for (int j = 0; j < 4; ++j) {
                a0[j]      = (__bf16)a00[j];
                a0[4 + j]  = (__bf16)a01[j];
                a0[8 + j]  = (__bf16)a02[j];
                a0[12 + j] = (__bf16)a03[j];
                a1[j]      = (__bf16)a10[j];
                a1[4 + j]  = (__bf16)a11[j];
                a1[8 + j]  = (__bf16)a12[j];
                a1[12 + j] = (__bf16)a13[j];
            }
            // Each B fragment feeds two WMMAs (tile 0 and tile 1).
            bf16x16 b0 = sB[(kt * 3 + 0) * 32 + lane];
            c00 = __builtin_amdgcn_wmma_f32_16x16x32_bf16(false, a0, false, b0,
                                                          (short)0, c00, false, false);
            c10 = __builtin_amdgcn_wmma_f32_16x16x32_bf16(false, a1, false, b0,
                                                          (short)0, c10, false, false);
            bf16x16 b1 = sB[(kt * 3 + 1) * 32 + lane];
            c01 = __builtin_amdgcn_wmma_f32_16x16x32_bf16(false, a0, false, b1,
                                                          (short)0, c01, false, false);
            c11 = __builtin_amdgcn_wmma_f32_16x16x32_bf16(false, a1, false, b1,
                                                          (short)0, c11, false, false);
            bf16x16 b2 = sB[(kt * 3 + 2) * 32 + lane];
            c02 = __builtin_amdgcn_wmma_f32_16x16x32_bf16(false, a0, false, b2,
                                                          (short)0, c02, false, false);
            c12 = __builtin_amdgcn_wmma_f32_16x16x32_bf16(false, a1, false, b2,
                                                          (short)0, c12, false, false);
        }

        // C/D 16x16 layout: VGPR r -> row (r + 8*hi), col = lidx.
#pragma unroll
        for (int r = 0; r < 8; ++r) {
            size_t row0 = (size_t)pair * 32 + r + hi * 8;
            float* o0 = emis + row0 * KK;
            o0[lidx]      = c00[r] + bi0;
            o0[16 + lidx] = c01[r] + bi1;
            o0[32 + lidx] = c02[r] + bi2;
            float* o1 = o0 + (size_t)16 * KK;
            o1[lidx]      = c10[r] + bi0;
            o1[16 + lidx] = c11[r] + bi1;
            o1[32 + lidx] = c12[r] + bi2;
        }
    }
}

// ---------------------------------------------------------------------------
// Phase 2: per-batch CRF forward recursion + numerator.
// One 64-thread block per batch column b. E = exp(transitions) precomputed
// once so each of the 511 sequential steps is a 48x48 FMA matvec + 48 exp +
// 48 log instead of 48x48 exps (transitions in [-0.1,0.1] -> global-max shift
// is numerically safe).
// ---------------------------------------------------------------------------
__global__ __launch_bounds__(64)
void crf_forward_kernel(const float* __restrict__ emis,
                        const float* __restrict__ startT,
                        const float* __restrict__ endT,
                        const float* __restrict__ trans,
                        const int* __restrict__ tags,
                        const uint8_t* __restrict__ mask,
                        float* __restrict__ llh)
{
    __shared__ float E[KK * KK];     // exp(transitions)
    __shared__ float alpha[KK];
    __shared__ float pp[KK];
    __shared__ float redf[64];
    __shared__ int   redi[64];

    const int b   = blockIdx.x;
    const int tid = threadIdx.x;

    for (int i = tid; i < KK * KK; i += 64) E[i] = __expf(trans[i]);
    if (tid < KK)
        alpha[tid] = startT[tid] + emis[(size_t)b * KK + tid];
    __syncthreads();

    for (int t = 1; t < TT; ++t) {
        // global max of alpha (broadcast LDS reads, conflict-free)
        float m = -3.402823466e38f;
        for (int i = 0; i < KK; ++i) m = fmaxf(m, alpha[i]);
        if (tid < KK) pp[tid] = __expf(alpha[tid] - m);
        __syncthreads();

        float nv = 0.0f;
        if (tid < KK) {
            float s = 0.0f;
            for (int i = 0; i < KK; ++i)
                s = fmaf(pp[i], E[i * KK + tid], s);
            nv = emis[((size_t)t * BB + b) * KK + tid] + m + __logf(s);
        }
        bool mk = mask[t * BB + b] != 0;
        __syncthreads();
        if (tid < KK && mk) alpha[tid] = nv;
        __syncthreads();
    }

    // denominator: logsumexp(alpha + endT)
    float m2 = -3.402823466e38f;
    for (int i = 0; i < KK; ++i) m2 = fmaxf(m2, alpha[i] + endT[i]);
    float s2 = 0.0f;
    for (int i = 0; i < KK; ++i) s2 += __expf(alpha[i] + endT[i] - m2);
    float denom = m2 + __logf(s2);

    // numerator: gathered path score (parallel over t, then reduce)
    float np = 0.0f;
    int cnt = 0;
    for (int t = tid; t < TT; t += 64) {
        int     tg = tags[t * BB + b];
        uint8_t mk = mask[t * BB + b];
        cnt += mk ? 1 : 0;
        if (t == 0) {
            np += startT[tg] + emis[(size_t)b * KK + tg];
        } else if (mk) {
            int tgp = tags[(t - 1) * BB + b];
            np += trans[tgp * KK + tg] + emis[((size_t)t * BB + b) * KK + tg];
        }
    }
    redf[tid] = np;
    redi[tid] = cnt;
    __syncthreads();
    if (tid == 0) {
        float num = 0.0f;
        int   nm  = 0;
        for (int i = 0; i < 64; ++i) { num += redf[i]; nm += redi[i]; }
        int seq_end = nm - 1;
        int last    = tags[seq_end * BB + b];
        num += endT[last];
        llh[b] = num - denom;
    }
}

// ---------------------------------------------------------------------------
// Phase 3: scalar reduction of 512 per-batch log-likelihoods -> d_out[0].
// ---------------------------------------------------------------------------
__global__ __launch_bounds__(256)
void final_sum_kernel(const float* __restrict__ llh, float* __restrict__ out)
{
    __shared__ float red[256];
    int tid = threadIdx.x;
    red[tid] = llh[tid] + llh[tid + 256];
    __syncthreads();
    for (int s = 128; s > 0; s >>= 1) {
        if (tid < s) red[tid] += red[tid + s];
        __syncthreads();
    }
    if (tid == 0) out[0] = red[0];
}

extern "C" void kernel_launch(void* const* d_in, const int* in_sizes, int n_in,
                              void* d_out, int out_size, void* d_ws, size_t ws_size,
                              hipStream_t stream)
{
    (void)in_sizes; (void)n_in; (void)out_size; (void)ws_size;
    const float*   hidden = (const float*)d_in[0];
    const float*   W      = (const float*)d_in[1];
    const float*   bias   = (const float*)d_in[2];
    const float*   startT = (const float*)d_in[3];
    const float*   endT   = (const float*)d_in[4];
    const float*   trans  = (const float*)d_in[5];
    const int*     tags   = (const int*)d_in[6];
    const uint8_t* mask   = (const uint8_t*)d_in[7];

    float* emis = (float*)d_ws;                                        // T*B*K f32 = 48 MB
    float* llh  = (float*)((char*)d_ws + (size_t)TT * BB * KK * sizeof(float)); // 512 f32

    emissions_gemm_kernel<<<1024, 256, 0, stream>>>(hidden, W, bias, emis);
    crf_forward_kernel<<<BB, 64, 0, stream>>>(emis, startT, endT, trans,
                                              tags, mask, llh);
    final_sum_kernel<<<1, 256, 0, stream>>>(llh, (float*)d_out);
}
